// QLenet_90220083020206
// MI455X (gfx1250) — 8-bit matrix operand order repaired and checked by reading, compiled, not run on hardware
//
#include <hip/hip_runtime.h>
#include <hip/hip_fp16.h>

typedef __attribute__((ext_vector_type(8))) int   v8i;
typedef __attribute__((ext_vector_type(8))) float v8f;

__device__ __forceinline__ float fq_f32(float x) {
    unsigned u  = __float_as_uint(x);
    unsigned au = u & 0x7fffffffu;
    unsigned qn = (u + 0x000fffffu + ((u >> 21) & 1u)) & 0xffe00000u;
    float sub = rintf(x * 65536.0f) * (1.0f / 65536.0f);
    float r = (au < 0x38800000u) ? sub : __uint_as_float(qn);
    float sinf = __uint_as_float((u & 0x80000000u) | 0x7f800000u);
    r = (au >= 0x47700000u) ? sinf : r;
    r = (au > 0x7f800000u) ? x : r;
    return r;
}

__device__ __forceinline__ unsigned char f32_to_bf8(float x) {
    __half h = __float2half(fq_f32(x));
    return (unsigned char)(__half_as_ushort(h) >> 8);
}

__global__ void k_quant_bf8(const float* __restrict__ in,
                            unsigned char* __restrict__ out, int n) {
    int i = blockIdx.x * blockDim.x + threadIdx.x;
    int s = gridDim.x * blockDim.x;
    for (; i < n; i += s) out[i] = f32_to_bf8(in[i]);
}

__global__ void k_zero_f32(float* p, int n) {
    int i = blockIdx.x * blockDim.x + threadIdx.x;
    if (i < n) p[i] = 0.f;
}

template <int C, int HW>
__global__ void __launch_bounds__(256)
k_bn_reduce(const float* __restrict__ x, float* __restrict__ sums, int Bn) {
    const int c = blockIdx.y;
    const int tot = Bn * HW;
    float s1 = 0.f, s2 = 0.f;
    int stride = gridDim.x * blockDim.x;
    for (int j = blockIdx.x * blockDim.x + threadIdx.x; j < tot; j += stride) {
        int b = j / HW, hw = j - b * HW;
        float v = x[((long)b * C + c) * HW + hw];
        s1 += v; s2 += v * v;
    }
    __shared__ float r1[256], r2[256];
    r1[threadIdx.x] = s1; r2[threadIdx.x] = s2;
    __syncthreads();
    #pragma unroll
    for (int s = 128; s > 0; s >>= 1) {
        if ((int)threadIdx.x < s) {
            r1[threadIdx.x] += r1[threadIdx.x + s];
            r2[threadIdx.x] += r2[threadIdx.x + s];
        }
        __syncthreads();
    }
    if (threadIdx.x == 0) {
        atomicAdd(&sums[c * 2 + 0], r1[0]);
        atomicAdd(&sums[c * 2 + 1], r2[0]);
    }
}

template <int C, int H, int W>
__global__ void __launch_bounds__(256)
k_bn_pool_q(const float* __restrict__ x, const float* __restrict__ sums,
            const float* __restrict__ g, const float* __restrict__ b,
            unsigned char* __restrict__ out, int Bn) {
    constexpr int PH = H / 2, PW = W / 2;
    int i = blockIdx.x * blockDim.x + threadIdx.x;
    if (i >= Bn * C * PH * PW) return;
    int px = i % PW;  int t = i / PW;
    int py = t % PH;  t /= PH;
    int c  = t % C;   int bb = t / C;
    float cnt = (float)Bn * H * W;
    float m  = sums[c * 2 + 0] / cnt;
    float vr = sums[c * 2 + 1] / cnt - m * m;
    float sc = rsqrtf(vr + 1e-5f) * g[c];
    float sh = b[c] - m * sc;
    const float* base = x + (((long)bb * C + c) * H + py * 2) * W + px * 2;
    float mx = 0.f;
    #pragma unroll
    for (int dy = 0; dy < 2; ++dy)
        #pragma unroll
        for (int dx = 0; dx < 2; ++dx) {
            float v = fq_f32(base[dy * W + dx] * sc + sh);
            mx = fmaxf(mx, fmaxf(v, 0.f));
        }
    out[i] = f32_to_bf8(mx);
}

#define MT 128
#define NT 16

template <int MODE, int N, int K, int Cin, int H, int W, int OH, int OW, int Cout>
__global__ void __launch_bounds__(256)
k_gemm_bf8(const unsigned char* __restrict__ Aq,
           const unsigned char* __restrict__ Wq,
           const float* __restrict__ bias,
           float* __restrict__ outF, unsigned char* __restrict__ outQ, int M) {
    __shared__ __align__(16) unsigned char As[2][MT * 64];
    __shared__ __align__(16) unsigned char Bs[2][NT * 64];
    const int tid  = threadIdx.x;
    const int lane = tid & 31;
    const int wave = tid >> 5;
    const int half = lane >> 4;
    const int ln   = lane & 15;
    const int m0   = blockIdx.x * MT;
    const int n0   = blockIdx.y * NT;
    constexpr int OHW = OH * OW;

    const unsigned char* cbase = nullptr;
    int crow = 0, ckbase = 0; bool cmok = false;
    if (MODE == 0) {
        crow   = tid >> 1;
        ckbase = (tid & 1) * 32;
        int m  = m0 + crow;
        cmok   = m < M;
        int bch = m / OHW, rem = m - bch * OHW;
        int oy = rem / OW, ox = rem - oy * OW;
        cbase = Aq + (((long)bch * Cin) * H + oy) * W + ox;
    }

    v8f acc = {0.f, 0.f, 0.f, 0.f, 0.f, 0.f, 0.f, 0.f};

    #pragma unroll
    for (int k0 = 0; k0 < K; k0 += 64) {
        const int p = (k0 >> 6) & 1;
        unsigned char* Ap = As[p];
        unsigned char* Bp = Bs[p];
        if (MODE == 0) {
            #pragma unroll
            for (int kk = 0; kk < 32; ++kk) {
                int k  = ckbase + kk;
                int kg = k0 + k;
                unsigned char v = 0;
                if (cmok && kg < K) {
                    int c  = kg / 25, r2 = kg - c * 25;
                    int fy = r2 / 5,  fx = r2 - fy * 5;
                    v = cbase[(c * H + fy) * W + fx];
                }
                Ap[crow * 64 + k] = v;
            }
            #pragma unroll
            for (int it = 0; it < 4; ++it) {
                int i = tid + 256 * it;
                int n = i >> 6, k = i & 63;
                int nn = n0 + n, kg = k0 + k;
                Bp[i] = (nn < N && kg < K) ? Wq[nn * K + kg] : (unsigned char)0;
            }
        } else {
            #pragma unroll
            for (int it = 0; it < 8; ++it) {
                int i = tid + 256 * it;
                int row = i >> 4, kd = (i & 15) << 2;
                int m = m0 + row, kg = k0 + kd;
                unsigned v = 0;
                if (m < M && kg < K) {
                    const unsigned char* src = Aq + (long)m * K + kg;
                    v = *(const unsigned*)src;
                    if (kg + 64 < K) __builtin_prefetch(src + 64, 0, 1);
                }
                *(unsigned*)&Ap[row * 64 + kd] = v;
            }
            {
                int n = tid >> 4, kd = (tid & 15) << 2;
                int nn = n0 + n, kg = k0 + kd;
                unsigned v = 0;
                if (nn < N && kg < K) v = *(const unsigned*)(Wq + nn * K + kg);
                *(unsigned*)&Bp[n * 64 + kd] = v;
            }
        }
        __syncthreads();

        v8i a, bf;
        const int arow = wave * 16 + ln;
        #pragma unroll
        for (int j = 0; j < 8; ++j) {
            int kb = ((j & 1) << 2) | (((j >> 1) & 1) << 4) | ((j >> 2) << 5) | (half << 3);
            a[j] = *(const int*)&Ap[arow * 64 + kb];
        }
        #pragma unroll
        for (int j = 0; j < 8; ++j) {
            int kb = ((j & 1) << 2) | (((j >> 1) & 1) << 4) | ((j >> 2) << 5) | (half << 3);
            bf[j] = *(const int*)&Bp[ln * 64 + kb];
        }
        acc = __builtin_amdgcn_wmma_f32_16x16x64_bf8_bf8(a, bf, (short)0, acc,
                                                         false, false);
    }

    const int n = n0 + ln;
    float bfq = 0.f;
    if (MODE != 0 && n < N) bfq = fq_f32(bias[n]);
    #pragma unroll
    for (int r = 0; r < 8; ++r) {
        int m = m0 + wave * 16 + r + 8 * half;
        if (m >= M || n >= N) continue;
        float y = acc[r];
        if (MODE == 0) {
            y = fq_f32(y);
            int bch = m / OHW, rem = m - bch * OHW;
            outF[((long)bch * Cout + n) * OHW + rem] = y;
        } else {
            y = fq_f32(y + bfq);
            if (MODE == 1) {
                outQ[(long)m * N + n] = f32_to_bf8(fmaxf(y, 0.f));
            } else {
                outF[(long)m * N + n] = y;
            }
        }
    }
}

extern "C" void kernel_launch(void* const* d_in, const int* in_sizes, int n_in,
                              void* d_out, int out_size, void* d_ws, size_t ws_size,
                              hipStream_t stream) {
    const float* x       = (const float*)d_in[0];
    const float* conv1_w = (const float*)d_in[1];
    const float* bn1_g   = (const float*)d_in[2];
    const float* bn1_b   = (const float*)d_in[3];
    const float* conv2_w = (const float*)d_in[4];
    const float* bn2_g   = (const float*)d_in[5];
    const float* bn2_b   = (const float*)d_in[6];
    const float* fc1_w   = (const float*)d_in[7];
    const float* fc1_b   = (const float*)d_in[8];
    const float* fc2_w   = (const float*)d_in[9];
    const float* fc2_b   = (const float*)d_in[10];
    const float* fc3_w   = (const float*)d_in[11];
    const float* fc3_b   = (const float*)d_in[12];

    const int B = 8192;
    unsigned char* ws = (unsigned char*)d_ws;
    size_t off = 0;
    auto alloc = [&](size_t bytes) {
        unsigned char* p = ws + off;
        off = (off + bytes + 255) & ~(size_t)255;
        return p;
    };

    unsigned char* xq    = alloc((size_t)B * 1024);
    unsigned char* w1q   = alloc(150);
    unsigned char* w2q   = alloc(2400);
    unsigned char* wf1q  = alloc(120 * 400);
    unsigned char* wf2q  = alloc(84 * 120);
    unsigned char* wf3q  = alloc(10 * 84);
    float* conv1o = (float*)alloc((size_t)B * 6 * 784 * 4);
    float* sums   = (float*)alloc(64 * 4);
    unsigned char* pool1 = alloc((size_t)B * 6 * 196);
    float* conv2o = conv1o;
    unsigned char* pool2 = alloc((size_t)B * 400);
    unsigned char* fc1o  = alloc((size_t)B * 120);
    unsigned char* fc2o  = alloc((size_t)B * 84);
    (void)ws_size; (void)in_sizes; (void)n_in; (void)out_size;

    k_quant_bf8<<<4096, 256, 0, stream>>>(x, xq, B * 1024);
    k_quant_bf8<<<1, 256, 0, stream>>>(conv1_w, w1q, 150);
    k_quant_bf8<<<10, 256, 0, stream>>>(conv2_w, w2q, 2400);
    k_quant_bf8<<<188, 256, 0, stream>>>(fc1_w, wf1q, 48000);
    k_quant_bf8<<<40, 256, 0, stream>>>(fc2_w, wf2q, 10080);
    k_quant_bf8<<<4, 256, 0, stream>>>(fc3_w, wf3q, 840);

    {
        int M = B * 784;
        dim3 grid((M + MT - 1) / MT, 1);
        k_gemm_bf8<0, 6, 25, 1, 32, 32, 28, 28, 6>
            <<<grid, 256, 0, stream>>>(xq, w1q, nullptr, conv1o, nullptr, M);
    }
    k_zero_f32<<<1, 64, 0, stream>>>(sums, 12);
    k_bn_reduce<6, 784><<<dim3(256, 6), 256, 0, stream>>>(conv1o, sums, B);
    {
        int total = B * 6 * 14 * 14;
        k_bn_pool_q<6, 28, 28><<<(total + 255) / 256, 256, 0, stream>>>(
            conv1o, sums, bn1_g, bn1_b, pool1, B);
    }

    {
        int M = B * 100;
        dim3 grid((M + MT - 1) / MT, 1);
        k_gemm_bf8<0, 16, 150, 6, 14, 14, 10, 10, 16>
            <<<grid, 256, 0, stream>>>(pool1, w2q, nullptr, conv2o, nullptr, M);
    }
    k_zero_f32<<<1, 64, 0, stream>>>(sums, 32);
    k_bn_reduce<16, 100><<<dim3(256, 16), 256, 0, stream>>>(conv2o, sums, B);
    {
        int total = B * 16 * 5 * 5;
        k_bn_pool_q<16, 10, 10><<<(total + 255) / 256, 256, 0, stream>>>(
            conv2o, sums, bn2_g, bn2_b, pool2, B);
    }

    {
        dim3 g1(B / MT, (120 + NT - 1) / NT);
        k_gemm_bf8<1, 120, 400, 1, 1, 1, 1, 1, 1>
            <<<g1, 256, 0, stream>>>(pool2, wf1q, fc1_b, nullptr, fc1o, B);
        dim3 g2(B / MT, (84 + NT - 1) / NT);
        k_gemm_bf8<1, 84, 120, 1, 1, 1, 1, 1, 1>
            <<<g2, 256, 0, stream>>>(fc1o, wf2q, fc2_b, nullptr, fc2o, B);
        dim3 g3(B / MT, 1);
        k_gemm_bf8<2, 10, 84, 1, 1, 1, 1, 1, 1>
            <<<g3, 256, 0, stream>>>(fc2o, wf3q, fc3_b, (float*)d_out, nullptr, B);
    }
}
